// GlobalAttention3d_9165460210374
// MI455X (gfx1250) — matrix operand order repaired and checked by reading, not compiled, not run on hardware
//
#include <hip/hip_runtime.h>

typedef _Float16 v8h  __attribute__((ext_vector_type(8)));
typedef _Float16 v16h __attribute__((ext_vector_type(16)));
typedef float    v8f  __attribute__((ext_vector_type(8)));

#define WMMA_F16(A,B,C) __builtin_amdgcn_wmma_f32_16x16x32_f16(false,(A),false,(B),(short)0,(C),false,false)

// ---- DPP 16-lane row reductions (pure VALU, no LDS) ----------------------
// xor-butterfly over masks {1,2,7,15}: quad_perm[1,0,3,2]=0xB1 (xor1),
// quad_perm[2,3,0,1]=0x4E (xor2), ROW_HALF_MIRROR=0x141 (xor7),
// ROW_MIRROR=0x140 (xor15). Union of partner sets covers all 16 lanes.
template<int CTRL>
__device__ __forceinline__ float dpp_mov_f(float x) {
  return __builtin_bit_cast(float,
      __builtin_amdgcn_update_dpp(0, __builtin_bit_cast(int, x), CTRL, 0xf, 0xf, true));
}
__device__ __forceinline__ float row_max16(float x) {
  x = fmaxf(x, dpp_mov_f<0xB1>(x));
  x = fmaxf(x, dpp_mov_f<0x4E>(x));
  x = fmaxf(x, dpp_mov_f<0x141>(x));
  x = fmaxf(x, dpp_mov_f<0x140>(x));
  return x;
}
__device__ __forceinline__ float row_sum16(float x) {
  x += dpp_mov_f<0xB1>(x);
  x += dpp_mov_f<0x4E>(x);
  x += dpp_mov_f<0x141>(x);
  x += dpp_mov_f<0x140>(x);
  return x;
}

// ---- elementwise f32 -> f16 ----------------------------------------------
__global__ __launch_bounds__(256) void cvt_f16_kernel(const float* __restrict__ in,
                                                      _Float16* __restrict__ out, int n) {
  int i = blockIdx.x * blockDim.x + threadIdx.x;
  if (i < n) out[i] = (_Float16)in[i];
}

// ---- GEMM: C[M,N] = A[M,K] * W[N,K]^T + bias; A,W f16; C f32 -------------
// 8 waves/block share one 64-col N-tile; weight block staged in LDS once.
// One wave computes a 16x64 tile (4 WMMA accumulators), K-loop step 32.
__global__ __launch_bounds__(256) void gemm_bias_kernel(
    const _Float16* __restrict__ A, const _Float16* __restrict__ W,
    const float* __restrict__ bias, float* __restrict__ Cout,
    int M, int N, int K) {
  __shared__ __align__(32) _Float16 wlds[2][64 * 32];  // 2 x 4KB weight tiles
  int wid  = (blockIdx.x * blockDim.x + threadIdx.x) >> 5;
  int lane = threadIdx.x & 31;
  int tm   = M >> 4;
  int mt   = wid % tm, nt = wid / tm;          // waves in a block share nt
  int l15 = lane & 15, hl = lane >> 4;
  int arow  = mt * 16 + l15;
  int abase = hl * 16;                         // A-layout: lanes 16-31 hold K=16..31
  int bbase = hl ? 16 : 0;                     // B-layout: lanes 16-31 hold K=16..31
  int t = threadIdx.x;
  int srr = t >> 2, sc8 = (t & 3) * 8;         // staging: 16B per thread
  v8f acc0 = {}, acc1 = {}, acc2 = {}, acc3 = {};
  for (int kb = 0; kb < K; kb += 32) {
    int p = (kb >> 5) & 1;
    // cooperative stage of W rows [nt*64, nt*64+64), cols [kb, kb+32)
    *(uint4*)&wlds[p][srr * 32 + sc8] =
        *(const uint4*)(W + (size_t)(nt * 64 + srr) * K + kb + sc8);
    __syncthreads();
    const _Float16* ap = A + (size_t)arow * K + kb + abase;
    v8h c1 = *(const v8h*)ap;
    v8h c2 = *(const v8h*)(ap + 8);
    v16h a;
    #pragma unroll
    for (int i = 0; i < 8; i++) { a[i] = c1[i]; a[8 + i] = c2[i]; }
    v16h b0 = *(const v16h*)&wlds[p][( 0 + l15) * 32 + bbase];
    v16h b1 = *(const v16h*)&wlds[p][(16 + l15) * 32 + bbase];
    v16h b2 = *(const v16h*)&wlds[p][(32 + l15) * 32 + bbase];
    v16h b3 = *(const v16h*)&wlds[p][(48 + l15) * 32 + bbase];
    acc0 = WMMA_F16(a, b0, acc0);
    acc1 = WMMA_F16(a, b1, acc1);
    acc2 = WMMA_F16(a, b2, acc2);
    acc3 = WMMA_F16(a, b3, acc3);
  }
  float bb0 = bias[nt * 64 +      l15];
  float bb1 = bias[nt * 64 + 16 + l15];
  float bb2 = bias[nt * 64 + 32 + l15];
  float bb3 = bias[nt * 64 + 48 + l15];
  #pragma unroll
  for (int r = 0; r < 8; r++) {
    int row = mt * 16 + r + hl * 8;            // C-layout rows
    float* out = Cout + (size_t)row * N + nt * 64 + l15;
    out[0]  = acc0[r] + bb0;
    out[16] = acc1[r] + bb1;
    out[32] = acc2[r] + bb2;
    out[48] = acc3[r] + bb3;
  }
}

// ---- Prep: l2norm q/k, scale+embed q, transpose v, emit f16 --------------
__global__ __launch_bounds__(256) void qkv_prep_kernel(
    const float* __restrict__ yq, const float* __restrict__ ykv,
    const float* __restrict__ qe, const float* __restrict__ temperature,
    _Float16* __restrict__ q16, _Float16* __restrict__ k16,
    _Float16* __restrict__ vt16) {
  int lin = blockIdx.x * blockDim.x + threadIdx.x;
  int d = lin & 31;
  int rest = lin >> 5;
  int n = rest & 2047;
  rest >>= 11;
  int h = rest % 12;
  int b = rest / 12;
  if (b >= 2) return;
  int bh = b * 12 + h;
  // q path
  float qv = yq[(size_t)(b * 2048 + n) * 384 + h * 32 + d];
  float ss = qv * qv;
  #pragma unroll
  for (int m = 1; m < 32; m <<= 1) ss += __shfl_xor(ss, m, 32);
  float qn = qv / fmaxf(sqrtf(ss), 1e-12f);
  float scale = log1pf(__expf(temperature[h])) * logf(2048.0f);
  float qo = (qn + qe[h * 32 + d]) * scale;
  q16[((size_t)bh * 2048 + n) * 32 + d] = (_Float16)qo;
  // k path
  float kv = ykv[(size_t)(b * 2048 + n) * 768 + h * 32 + d];
  float ks = kv * kv;
  #pragma unroll
  for (int m = 1; m < 32; m <<= 1) ks += __shfl_xor(ks, m, 32);
  float kn = kv / fmaxf(sqrtf(ks), 1e-12f);
  k16[((size_t)bh * 2048 + n) * 32 + d] = (_Float16)kn;
  // v path (transposed: [bh][d][n])
  float vv = ykv[(size_t)(b * 2048 + n) * 768 + 384 + h * 32 + d];
  vt16[((size_t)bh * 32 + d) * 2048 + n] = (_Float16)vv;
}

// ---- Flash attention -----------------------------------------------------
// 8 waves/block handle 8 consecutive 16-query tiles of the SAME (b,h)
// (128 tiles/head, 8 | 128 -> no block straddles a head). K/V 32-key blocks
// are staged in LDS once per block and shared by all 8 waves.
__global__ __launch_bounds__(256) void attn_kernel(
    const _Float16* __restrict__ q16, const _Float16* __restrict__ k16,
    const _Float16* __restrict__ vt16, const float* __restrict__ pos_bias,
    _Float16* __restrict__ att16) {
  __shared__ __align__(32) _Float16 kbuf[2][32 * 32];  // 2 x 2KB K tiles
  __shared__ __align__(32) _Float16 vbuf[2][32 * 32];  // 2 x 2KB V tiles
  __shared__ __align__(32) _Float16 plds[8][16 * 32];  // per-wave P bounce
  int wid  = (blockIdx.x * blockDim.x + threadIdx.x) >> 5;
  int lane = threadIdx.x & 31;
  int wslot = threadIdx.x >> 5;
  int bh = wid >> 7;          // 128 q-tiles per (b,h); uniform across block
  int qt = wid & 127;
  int h = bh % 12, b = bh / 12;
  int l15 = lane & 15, hl = lane >> 4;
  int abase = hl * 16;        // A-layout half-wave K offset
  int bbase = hl ? 16 : 0;    // B-layout half-wave K offset
  int t = threadIdx.x;
  int srow = t >> 3, sc4 = (t & 7) * 4;        // staging: 8B per thread

  // Q tile in A-matrix register layout (resident whole kernel)
  const _Float16* qrow = q16 + ((size_t)bh * 2048 + qt * 16 + l15) * 32;
  v16h aq;
  {
    v8h c1 = *(const v8h*)(qrow + abase);
    v8h c2 = *(const v8h*)(qrow + abase + 8);
    #pragma unroll
    for (int i = 0; i < 8; i++) { aq[i] = c1[i]; aq[8 + i] = c2[i]; }
  }

  v8f o0 = {}, o1 = {};
  float lsum[8], mrun[8];
  #pragma unroll
  for (int r = 0; r < 8; r++) { lsum[r] = 0.0f; mrun[r] = -1e30f; }
  const float* pb = pos_bias + h * 2048;
  _Float16* pl = &plds[wslot][0];
  const _Float16* ksrc = k16  + (size_t)bh * 2048 * 32;
  const _Float16* vsrc = vt16 + (size_t)bh * 32 * 2048;

  for (int kb = 0; kb < 2048; kb += 32) {
    int p = (kb >> 5) & 1;
    // cooperative stage: K rows [kb,kb+32) (32x32 halves), V rows d=0..31
    *(uint2*)&kbuf[p][srow * 32 + sc4] =
        *(const uint2*)(ksrc + (size_t)(kb + srow) * 32 + sc4);
    *(uint2*)&vbuf[p][srow * 32 + sc4] =
        *(const uint2*)(vsrc + (size_t)srow * 2048 + kb + sc4);
    __syncthreads();
    // K tiles in B-matrix layout from LDS
    v16h b0 = *(const v16h*)&kbuf[p][( 0 + l15) * 32 + bbase];
    v16h b1 = *(const v16h*)&kbuf[p][(16 + l15) * 32 + bbase];
    v8f zc = {};
    v8f s0 = WMMA_F16(aq, b0, zc);
    v8f s1 = WMMA_F16(aq, b1, zc);
    float pb0 = pb[kb + l15], pb1 = pb[kb + 16 + l15];
    #pragma unroll
    for (int r = 0; r < 8; r++) {
      s0[r] += pb0; s1[r] += pb1;
      float tm = row_max16(fmaxf(s0[r], s1[r]));   // DPP butterfly, no LDS
      float mn = fmaxf(mrun[r], tm);
      float f  = __expf(mrun[r] - mn);
      mrun[r]  = mn;
      o0[r] *= f; o1[r] *= f; lsum[r] *= f;
      float p0 = __expf(s0[r] - mn);
      float p1 = __expf(s1[r] - mn);
      lsum[r] += p0 + p1;
      int m0 = r + hl * 8;                       // C-layout row of this VGPR
      pl[m0 * 32 +      l15] = (_Float16)p0;     // row-major 16x32 P in LDS
      pl[m0 * 32 + 16 + l15] = (_Float16)p1;
    }
    // Reload P in A-matrix layout (DS ops are in-order within a wave)
    v16h pa;
    {
      const _Float16* pp = pl + l15 * 32 + abase;
      v8h c1 = *(const v8h*)pp;
      v8h c2 = *(const v8h*)(pp + 8);
      #pragma unroll
      for (int i = 0; i < 8; i++) { pa[i] = c1[i]; pa[8 + i] = c2[i]; }
    }
    // V in B-matrix layout from LDS (vt16 is key-contiguous per d-row)
    v16h bv0 = *(const v16h*)&vbuf[p][( 0 + l15) * 32 + bbase];
    v16h bv1 = *(const v16h*)&vbuf[p][(16 + l15) * 32 + bbase];
    o0 = WMMA_F16(pa, bv0, o0);
    o1 = WMMA_F16(pa, bv1, o1);
  }
  #pragma unroll
  for (int r = 0; r < 8; r++) {
    float inv = 1.0f / row_sum16(lsum[r]);       // DPP sum across the 16-lane row
    int n = qt * 16 + r + hl * 8;
    _Float16* out = att16 + ((size_t)b * 2048 + n) * 384 + h * 32;
    out[l15]      = (_Float16)(o0[r] * inv);
    out[16 + l15] = (_Float16)(o1[r] * inv);
  }
}

// ---- launch --------------------------------------------------------------
extern "C" void kernel_launch(void* const* d_in, const int* in_sizes, int n_in,
                              void* d_out, int out_size, void* d_ws, size_t ws_size,
                              hipStream_t stream) {
  const float* x      = (const float*)d_in[0];
  const float* q_w    = (const float*)d_in[1];
  const float* q_b    = (const float*)d_in[2];
  const float* kv_w   = (const float*)d_in[3];
  const float* kv_b   = (const float*)d_in[4];
  const float* qe     = (const float*)d_in[5];
  const float* temp   = (const float*)d_in[6];
  const float* posb   = (const float*)d_in[7];
  const float* proj_w = (const float*)d_in[8];
  const float* proj_b = (const float*)d_in[9];

  char* base = (char*)d_ws;
  size_t off = 0;
  auto carve = [&](size_t bytes) -> char* {
    char* p = base + off;
    off += (bytes + 255) & ~(size_t)255;
    return p;
  };
  float*     yq    = (float*)    carve((size_t)4096 * 384 * 4);   // raw q
  float*     ykv   = (float*)    carve((size_t)4096 * 768 * 4);   // raw kv
  _Float16*  q16   = (_Float16*) carve((size_t)24 * 2048 * 32 * 2);
  _Float16*  k16   = (_Float16*) carve((size_t)24 * 2048 * 32 * 2);
  _Float16*  vt16  = (_Float16*) carve((size_t)24 * 32 * 2048 * 2);
  _Float16*  a16   = (_Float16*) carve((size_t)4096 * 384 * 2);   // attn out (f16)
  _Float16*  x16   = (_Float16*) carve((size_t)4096 * 384 * 2);
  _Float16*  qw16  = (_Float16*) carve((size_t)384 * 384 * 2);
  _Float16*  kvw16 = (_Float16*) carve((size_t)768 * 384 * 2);
  _Float16*  pw16  = (_Float16*) carve((size_t)384 * 384 * 2);

  // Pre-convert activations & weights to f16 (GEMMs become pure b128 + WMMA)
  cvt_f16_kernel<<<6144, 256, 0, stream>>>(x, x16, 4096 * 384);
  cvt_f16_kernel<<<576,  256, 0, stream>>>(q_w, qw16, 384 * 384);
  cvt_f16_kernel<<<1152, 256, 0, stream>>>(kv_w, kvw16, 768 * 384);
  cvt_f16_kernel<<<576,  256, 0, stream>>>(proj_w, pw16, 384 * 384);

  // GEMM1: yq = x @ q_w^T + q_b           (M=4096,N=384,K=384) -> 1536 waves
  gemm_bias_kernel<<<192, 256, 0, stream>>>(x16, qw16, q_b, yq, 4096, 384, 384);
  // GEMM2: ykv = x @ kv_w^T + kv_b        (M=4096,N=768,K=384) -> 3072 waves
  gemm_bias_kernel<<<384, 256, 0, stream>>>(x16, kvw16, kv_b, ykv, 4096, 768, 384);
  // Prep: 2*12*2048*32 = 1,572,864 threads
  qkv_prep_kernel<<<6144, 256, 0, stream>>>(yq, ykv, qe, temp, q16, k16, vt16);
  // Attention: 24 heads * 128 q-tiles = 3072 waves, 8 waves/block share K/V
  attn_kernel<<<384, 256, 0, stream>>>(q16, k16, vt16, posb, a16);
  // GEMM3: out = attn @ proj_w^T + proj_b
  gemm_bias_kernel<<<192, 256, 0, stream>>>(a16, pw16, proj_b, (float*)d_out,
                                            4096, 384, 384);
}
